// MetaLayerModel_70884140253680
// MI455X (gfx1250) — matrix operand order repaired and checked by reading, not compiled, not run on hardware
//
#include <hip/hip_runtime.h>

#define LEAK 0.1f
#define EPSV 1e-5f
#define CH 512

typedef __attribute__((ext_vector_type(16))) __bf16 v16bf;
typedef __attribute__((ext_vector_type(8)))  float v8f;
typedef __attribute__((ext_vector_type(4)))  float v4f;

__device__ __forceinline__ unsigned short f2bf(float f) {
    unsigned int u = __float_as_uint(f);
    unsigned int r = u + 0x7FFFu + ((u >> 16) & 1u);   // round-to-nearest-even
    return (unsigned short)(r >> 16);
}

// ---------------------------------------------------------------- utilities
__global__ void k_zero(float* __restrict__ p, long long n) {
    long long i = (long long)blockIdx.x * blockDim.x + threadIdx.x;
    long long step = (long long)gridDim.x * blockDim.x;
    for (; i < n; i += step) p[i] = 0.f;
}

__global__ void k_count(const int* __restrict__ col, float* __restrict__ cnt, int E) {
    int i = blockIdx.x * blockDim.x + threadIdx.x;
    if (i < E) atomicAdd(&cnt[col[i]], 1.f);
}

__global__ void k_norm(float* __restrict__ agg, const float* __restrict__ cnt, long long n) {
    long long i = (long long)blockIdx.x * blockDim.x + threadIdx.x;
    long long step = (long long)gridDim.x * blockDim.x;
    for (; i < n; i += step) agg[i] /= fmaxf(cnt[i >> 6], 1.f);
}

// Zero-pad edge features [E,19] -> [E,32]
__global__ void k_pad_edge(const float* __restrict__ e, float* __restrict__ epad, int E) {
    int i = blockIdx.x * blockDim.x + threadIdx.x;
    if (i >= E * 32) return;
    int r = i >> 5, j = i & 31;
    epad[i] = (j < 19) ? e[(size_t)r * 19 + j] : 0.f;
}

// Column mean/var stats over the virtual matrix cat(s0[i0], s1[i1], s2) : [R, d0+d1+d2]
__global__ void k_colstats(const float* __restrict__ s0, const int* __restrict__ i0, int d0,
                           const float* __restrict__ s1, const int* __restrict__ i1, int d1,
                           const float* __restrict__ s2, int d2,
                           int R, float* __restrict__ sums, float* __restrict__ sumsq) {
    int K = d0 + d1 + d2;
    int tid = blockIdx.x * blockDim.x + threadIdx.x;
    if (tid >= K * CH) return;
    int k = tid % K;
    int chunk = tid / K;
    float s = 0.f, q = 0.f;
    for (int r = chunk; r < R; r += CH) {
        float v;
        if (k < d0)           v = s0[(size_t)(i0 ? i0[r] : r) * d0 + k];
        else if (k < d0 + d1) v = s1[(size_t)(i1 ? i1[r] : r) * d1 + (k - d0)];
        else                  v = s2[(size_t)r * d2 + (k - d0 - d1)];
        s += v; q += v * v;
    }
    atomicAdd(&sums[k], s);
    atomicAdd(&sumsq[k], q);
}

// Standalone BN apply (outer per-layer BN on x)
__global__ void k_apply_bn(const float* __restrict__ X, float* __restrict__ Y,
                           long long n, int D, float invR,
                           const float* __restrict__ sums, const float* __restrict__ sumsq,
                           const float* __restrict__ g, const float* __restrict__ b) {
    long long i = (long long)blockIdx.x * blockDim.x + threadIdx.x;
    long long step = (long long)gridDim.x * blockDim.x;
    for (; i < n; i += step) {
        int k = (int)(i % D);
        float mean = sums[k] * invR;
        float var  = sumsq[k] * invR - mean * mean;
        float sc   = g[k] * rsqrtf(var + EPSV);
        Y[i] = (X[i] - mean) * sc + b[k];
    }
}

// Fold BN(scale,shift) into W ([64 out, K in] row-major), pack bf16 B-operand:
// element index = ((kb*4 + c)*32 + lane)*16 + t,
// k = kb*32 + (t<8 ? 0 : 16) + (lane<16 ? 0 : 8) + (t&7) ; n = c*16 + (lane&15). Rows k>=K are zero.
// Also folded bias' = b + sum_k shift_k * W[n][k].
__global__ void k_fold(const float* __restrict__ W, const float* __restrict__ b,
                       const float* __restrict__ bng, const float* __restrict__ bnb,
                       const float* __restrict__ sums, const float* __restrict__ sumsq,
                       int K, int KB, float invR,
                       unsigned short* __restrict__ Bp, float* __restrict__ biasp) {
    int total = KB * 4 * 32 * 16;
    int tid = blockIdx.x * blockDim.x + threadIdx.x;
    if (tid < total) {
        int t    = tid & 15;
        int lane = (tid >> 4) & 31;
        int c    = (tid >> 9) & 3;
        int kb   = tid >> 11;
        int k = kb * 32 + ((t < 8) ? 0 : 16) + ((lane < 16) ? 0 : 8) + (t & 7);
        int n = c * 16 + (lane & 15);
        float val = 0.f;
        if (k < K) {
            float mean = sums[k] * invR;
            float var  = sumsq[k] * invR - mean * mean;
            float sc   = bng[k] * rsqrtf(var + EPSV);
            val = W[(size_t)n * K + k] * sc;
        }
        Bp[tid] = f2bf(val);
    } else if (tid < total + 64) {
        int n = tid - total;
        float acc = b[n];
        for (int k = 0; k < K; ++k) {
            float mean = sums[k] * invR;
            float var  = sumsq[k] * invR - mean * mean;
            float sc   = bng[k] * rsqrtf(var + EPSV);
            float sh   = bnb[k] - mean * sc;
            acc += sh * W[(size_t)n * K + k];
        }
        biasp[n] = acc;
    }
}

// Up to 4 concatenated row-segments; every dim is a multiple of 16 and
// sum(dim) == KB*32, so every 16-wide window [kb*32+h*16, +16) lies inside
// exactly one segment -> segment selection is wave-uniform (scalar branch).
struct GemmSegs {
    const float* p[4];
    const int*   idx[4];
    int dim[4];
    int st[4];   // row stride (0 for the shared zero-pad segment)
};

// Fused gather/concat -> bf16 WMMA GEMM:  Y[R,64] = A @ Bp + biasp
// One wave per 16-row x 64-col tile; 8 waves (256 threads) per block.
// A-operand 16-bit layout (ISA 7.12.2): lane<16 holds K {0-7,16-23}+kb*32,
// lane>=16 holds K {8-15,24-31}+kb*32 for row M = lane&15.
__global__ __launch_bounds__(256)
void k_gemm_wmma(GemmSegs g, const unsigned short* __restrict__ Bp,
                 const float* __restrict__ biasp,
                 int R, int KB, float* __restrict__ Y,
                 const int* __restrict__ scat, int leaky) {
    const int lane = threadIdx.x & 31;
    const int wave = threadIdx.x >> 5;
    const int tileRow = (blockIdx.x * 8 + wave) * 16;
    const int m = lane & 15;
    int rowc = tileRow + m;
    if (rowc > R - 1) rowc = R - 1;   // clamp: loads stay valid, stores masked below

    const float* P0 = g.p[0] + (size_t)(g.idx[0] ? g.idx[0][rowc] : rowc) * g.st[0];
    const float* P1 = g.p[1] + (size_t)(g.idx[1] ? g.idx[1][rowc] : rowc) * g.st[1];
    const float* P2 = g.p[2] + (size_t)(g.idx[2] ? g.idx[2][rowc] : rowc) * g.st[2];
    const float* P3 = g.p[3] + (size_t)(g.idx[3] ? g.idx[3][rowc] : rowc) * g.st[3];
    const int c1 = g.dim[0], c2 = c1 + g.dim[1], c3 = c2 + g.dim[2];
    const int koff = (lane < 16) ? 0 : 8;

    v8f acc[4] = {};
    const v16bf* Bv = (const v16bf*)Bp;

    for (int kb = 0; kb < KB; ++kb) {
        v16bf A;
        #pragma unroll
        for (int h = 0; h < 2; ++h) {
            const int kw = kb * 32 + h * 16;       // SCALAR: uniform across wave
            // uniform segment select (s_cmp/s_cselect), then per-lane half offset
            const float* qb = (kw < c1) ? P0 + kw
                            : (kw < c2) ? P1 + (kw - c1)
                            : (kw < c3) ? P2 + (kw - c2)
                            :             P3 + (kw - c3);
            const float* q = qb + koff;
            v4f lo = *(const v4f*)q;
            v4f hi = *(const v4f*)(q + 4);
            A[h * 8 + 0] = (__bf16)lo.x; A[h * 8 + 1] = (__bf16)lo.y;
            A[h * 8 + 2] = (__bf16)lo.z; A[h * 8 + 3] = (__bf16)lo.w;
            A[h * 8 + 4] = (__bf16)hi.x; A[h * 8 + 5] = (__bf16)hi.y;
            A[h * 8 + 6] = (__bf16)hi.z; A[h * 8 + 7] = (__bf16)hi.w;
        }
        // preload all 4 B chunks, then issue 4 WMMAs back-to-back
        v16bf b0 = Bv[(size_t)(kb * 4 + 0) * 32 + lane];
        v16bf b1 = Bv[(size_t)(kb * 4 + 1) * 32 + lane];
        v16bf b2 = Bv[(size_t)(kb * 4 + 2) * 32 + lane];
        v16bf b3 = Bv[(size_t)(kb * 4 + 3) * 32 + lane];
        acc[0] = __builtin_amdgcn_wmma_f32_16x16x32_bf16(false, A, false, b0, (short)0, acc[0], false, false);
        acc[1] = __builtin_amdgcn_wmma_f32_16x16x32_bf16(false, A, false, b1, (short)0, acc[1], false, false);
        acc[2] = __builtin_amdgcn_wmma_f32_16x16x32_bf16(false, A, false, b2, (short)0, acc[2], false, false);
        acc[3] = __builtin_amdgcn_wmma_f32_16x16x32_bf16(false, A, false, b3, (short)0, acc[3], false, false);
    }

    // D tile f32 layout: lane<16 -> M=v, N=lane ; lane>=16 -> M=8+v, N=lane-16
    const int n0 = lane & 15;
    const int mb = (lane < 16) ? 0 : 8;
    #pragma unroll
    for (int c = 0; c < 4; ++c) {
        float bias = biasp[c * 16 + n0];
        #pragma unroll
        for (int v = 0; v < 8; ++v) {
            int r = tileRow + mb + v;
            if (r >= R) continue;
            float val = acc[c][v] + bias;
            if (leaky) val = (val >= 0.f) ? val : LEAK * val;
            int n = c * 16 + n0;
            if (scat) atomicAdd(&Y[(size_t)scat[r] * 64 + n], val);
            else      Y[(size_t)r * 64 + n] = val;
        }
    }
}

// Final heads: node_pred[N,2], edge_pred[E,2], copy x[N,64]
__global__ void k_heads(const float* __restrict__ x, const float* __restrict__ e,
                        const float* __restrict__ npW, const float* __restrict__ npb,
                        const float* __restrict__ epW, const float* __restrict__ epb,
                        float* __restrict__ out, int N, int E) {
    long long i = (long long)blockIdx.x * blockDim.x + threadIdx.x;
    long long nE = (long long)N + E;
    if (i < N) {
        #pragma unroll
        for (int c = 0; c < 2; ++c) {
            float acc = npb[c];
            for (int j = 0; j < 64; ++j) acc += x[i * 64 + j] * npW[c * 64 + j];
            out[i * 2 + c] = acc;
        }
    } else if (i < nE) {
        long long r = i - N;
        #pragma unroll
        for (int c = 0; c < 2; ++c) {
            float acc = epb[c];
            for (int j = 0; j < 64; ++j) acc += e[r * 64 + j] * epW[c * 64 + j];
            out[(long long)2 * N + r * 2 + c] = acc;
        }
    } else if (i < nE + (long long)N * 64) {
        long long j = i - nE;
        out[(long long)2 * N + (long long)2 * E + j] = x[j];
    }
}

// ---------------------------------------------------------------- host
extern "C" void kernel_launch(void* const* d_in, const int* in_sizes, int n_in,
                              void* d_out, int out_size, void* d_ws, size_t ws_size,
                              hipStream_t stream) {
    (void)in_sizes; (void)n_in; (void)out_size; (void)ws_size;
    const int N = 50000, E = 400000;
    const float* nodef = (const float*)d_in[0];
    const int*   eidx  = (const int*)d_in[1];     // [2,E]: row then col
    const float* edgef = (const float*)d_in[2];
    const int* row = eidx;
    const int* col = eidx + E;

    // params: pytree flatten order (sorted dict keys)
    int p = 4;
    auto F = [&](int i) { return (const float*)d_in[i]; };
    const float* epW = F(p++); const float* epb = F(p++);
    struct Mlp { const float *W[3], *b[3], *bnb[3], *bng[3]; };
    struct Layer { const float *bnb, *bng; Mlp mlp[3]; } L[3];
    for (int l = 0; l < 3; ++l) {
        L[l].bnb = F(p++); L[l].bng = F(p++);
        for (int mi = 0; mi < 3; ++mi) {
            Mlp& M = L[l].mlp[mi];
            for (int i = 0; i < 3; ++i) M.W[i]   = F(p++);
            for (int i = 0; i < 3; ++i) M.b[i]   = F(p++);
            for (int i = 0; i < 3; ++i) M.bnb[i] = F(p++);
            for (int i = 0; i < 3; ++i) M.bng[i] = F(p++);
        }
    }
    const float* npW = F(p++); const float* npb = F(p++);

    // workspace carve-up
    char* w = (char*)d_ws;
    auto alloc = [&](size_t bytes) -> void* {
        void* r = (void*)w; w += (bytes + 255) & ~(size_t)255; return r;
    };
    float* xbn   = (float*)alloc((size_t)N * 64 * 4);
    float* xnew  = (float*)alloc((size_t)N * 64 * 4);
    float* ebuf  = (float*)alloc((size_t)E * 64 * 4);
    float* tbuf  = (float*)alloc((size_t)E * 64 * 4);
    float* agg   = (float*)alloc((size_t)N * 64 * 4);
    float* epad  = (float*)alloc((size_t)E * 32 * 4);
    float* cnt   = (float*)alloc((size_t)N * 4);
    float* sums  = (float*)alloc(256 * 4);
    float* sumsq = (float*)alloc(256 * 4);
    unsigned short* Wp = (unsigned short*)alloc((size_t)8 * 4 * 32 * 16 * 2);
    float* biasp = (float*)alloc(64 * 4);
    float* zbuf  = (float*)alloc(64 * 4);

    auto zero = [&](float* ptr, long long n) {
        long long blocks = (n + 255) / 256; if (blocks > 4096) blocks = 4096;
        k_zero<<<(int)blocks, 256, 0, stream>>>(ptr, n);
    };
    auto stats = [&](const float* s0, const int* i0, int d0,
                     const float* s1, const int* i1, int d1,
                     const float* s2, int d2, int R) {
        zero(sums, 256); zero(sumsq, 256);
        int K = d0 + d1 + d2;
        int total = K * CH;
        k_colstats<<<(total + 255) / 256, 256, 0, stream>>>(
            s0, i0, d0, s1, i1, d1, s2, d2, R, sums, sumsq);
    };
    // logical stats sources (true concat) + padded gemm segments (multiples of 16)
    auto stage = [&](const Mlp& M, int si,
                     const float* s0, const int* i0, int d0,
                     const float* s1, const int* i1, int d1,
                     const float* s2, int d2,
                     int R, const GemmSegs& gs, float* Y, const int* scat, int leaky) {
        int K = d0 + d1 + d2;
        int Kp = gs.dim[0] + gs.dim[1] + gs.dim[2] + gs.dim[3];
        int KB = Kp / 32;
        stats(s0, i0, d0, s1, i1, d1, s2, d2, R);
        int ftotal = KB * 4 * 32 * 16 + 64;
        k_fold<<<(ftotal + 255) / 256, 256, 0, stream>>>(
            M.W[si], M.b[si], M.bng[si], M.bnb[si],
            sums, sumsq, K, KB, 1.f / R, Wp, biasp);
        k_gemm_wmma<<<(R + 127) / 128, 256, 0, stream>>>(
            gs, Wp, biasp, R, KB, Y, scat, leaky);
    };

    zero(zbuf, 64);
    zero(cnt, N);
    k_count<<<(E + 255) / 256, 256, 0, stream>>>(col, cnt, E);
    k_pad_edge<<<(E * 32 + 255) / 256, 256, 0, stream>>>(edgef, epad, E);

    const float* x = nodef;
    const float* e = edgef;
    for (int l = 0; l < 3; ++l) {
        int ni = l ? 64 : 16, ei = l ? 64 : 19;
        // outer BN on x
        stats(x, nullptr, ni, nullptr, nullptr, 0, nullptr, 0, N);
        k_apply_bn<<<2048, 256, 0, stream>>>(x, xbn, (long long)N * ni, ni, 1.f / N,
                                             sums, sumsq, L[l].bng, L[l].bnb);

        GemmSegs segE1 = (l == 0)
            ? GemmSegs{ {xbn, xbn, epad, zbuf}, {row, col, nullptr, nullptr},
                        {16, 16, 32, 0}, {16, 16, 32, 0} }            // Kp=64, KB=2
            : GemmSegs{ {xbn, xbn, ebuf, zbuf}, {row, col, nullptr, nullptr},
                        {64, 64, 64, 0}, {64, 64, 64, 0} };           // Kp=192, KB=6
        GemmSegs seg64 = { {tbuf, zbuf, zbuf, zbuf}, {nullptr, nullptr, nullptr, nullptr},
                           {64, 0, 0, 0}, {64, 0, 0, 0} };            // Kp=64, KB=2
        GemmSegs segN1 = (l == 0)
            ? GemmSegs{ {xbn, ebuf, zbuf, zbuf}, {row, nullptr, nullptr, nullptr},
                        {16, 64, 16, 0}, {16, 64, 0, 0} }             // Kp=96, KB=3
            : GemmSegs{ {xbn, ebuf, zbuf, zbuf}, {row, nullptr, nullptr, nullptr},
                        {64, 64, 0, 0}, {64, 64, 0, 0} };             // Kp=128, KB=4
        GemmSegs segN2 = (l == 0)
            ? GemmSegs{ {xbn, agg, zbuf, zbuf}, {nullptr, nullptr, nullptr, nullptr},
                        {16, 64, 16, 0}, {16, 64, 0, 0} }
            : GemmSegs{ {xbn, agg, zbuf, zbuf}, {nullptr, nullptr, nullptr, nullptr},
                        {64, 64, 0, 0}, {64, 64, 0, 0} };

        // edge MLP3: cat(x[row], x[col], e) -> t -> t -> ebuf
        stage(L[l].mlp[0], 0, xbn, row, ni, xbn, col, ni, e, ei, E, segE1, tbuf, nullptr, 1);
        stage(L[l].mlp[0], 1, tbuf, nullptr, 64, nullptr, nullptr, 0, nullptr, 0, E, seg64, tbuf, nullptr, 1);
        stage(L[l].mlp[0], 2, tbuf, nullptr, 64, nullptr, nullptr, 0, nullptr, 0, E, seg64, ebuf, nullptr, 0);
        // node MLP1: cat(x[row], e) -> t -> t -> scatter-add into agg[col]
        stage(L[l].mlp[1], 0, xbn, row, ni, ebuf, nullptr, 64, nullptr, 0, E, segN1, tbuf, nullptr, 1);
        stage(L[l].mlp[1], 1, tbuf, nullptr, 64, nullptr, nullptr, 0, nullptr, 0, E, seg64, tbuf, nullptr, 1);
        zero(agg, (long long)N * 64);
        stage(L[l].mlp[1], 2, tbuf, nullptr, 64, nullptr, nullptr, 0, nullptr, 0, E, seg64, agg, col, 0);
        k_norm<<<2048, 256, 0, stream>>>(agg, cnt, (long long)N * 64);
        // node MLP2: cat(x, agg) -> t -> t -> xnew
        stage(L[l].mlp[2], 0, xbn, nullptr, ni, agg, nullptr, 64, nullptr, 0, N, segN2, tbuf, nullptr, 1);
        stage(L[l].mlp[2], 1, tbuf, nullptr, 64, nullptr, nullptr, 0, nullptr, 0, N, seg64, tbuf, nullptr, 1);
        stage(L[l].mlp[2], 2, tbuf, nullptr, 64, nullptr, nullptr, 0, nullptr, 0, N, seg64, xnew, nullptr, 0);
        x = xnew; e = ebuf;
    }

    long long htotal = (long long)N + E + (long long)N * 64;
    k_heads<<<(int)((htotal + 255) / 256), 256, 0, stream>>>(
        xnew, ebuf, npW, npb, epW, epb, (float*)d_out, N, E);
}
